// WSDDN_5059471475226
// MI455X (gfx1250) — compile-verified, matrix operand order checked by reading, not run on hardware
//
#include <hip/hip_runtime.h>

// ---------- types ----------
typedef __attribute__((ext_vector_type(16))) __bf16 v16bf;
typedef __attribute__((ext_vector_type(8)))  float  v8f;

union FragBF16 { v16bf v; unsigned int u[8]; };

// ---------- bf16 helpers (raw-bit storage, RNE convert) ----------
__device__ __forceinline__ unsigned short f2bf(float f) {
    unsigned int u = __float_as_uint(f);
    u += 0x7FFFu + ((u >> 16) & 1u);
    return (unsigned short)(u >> 16);
}
__device__ __forceinline__ float bf2f(unsigned short h) {
    return __uint_as_float(((unsigned int)h) << 16);
}

// ---------- elementwise f32 -> bf16 ----------
__global__ void f32_to_bf16_k(const float* __restrict__ in, unsigned short* __restrict__ out, int n) {
    int id = blockIdx.x * blockDim.x + threadIdx.x;
    if (id < n) out[id] = f2bf(in[id]);
}

// ---------- conv weight [N=Cout][K=Cin*9] f32 -> bf16 [N][Kp] (K zero-padded) ----------
__global__ void conv_w_to_bf16_k(const float* __restrict__ w, unsigned short* __restrict__ wb,
                                 int K, int Kp, int N) {
    int id = blockIdx.x * blockDim.x + threadIdx.x;
    if (id >= Kp * N) return;
    int n = id / Kp, k = id % Kp;
    float v = (k < K) ? w[(size_t)n * K + k] : 0.0f;
    wb[(size_t)n * Kp + k] = f2bf(v);
}

// ---------- tiled transpose-convert: f32 [K][N] -> bf16 [N][K] (K,N multiples of 32) ----------
__global__ void transpose_f32_to_bf16_k(const float* __restrict__ in, unsigned short* __restrict__ out,
                                        int K, int N) {
    __shared__ float tile[32][33];
    const int kb = blockIdx.x * 32, nb = blockIdx.y * 32;
    const int tx = threadIdx.x, ty = threadIdx.y;   // block (32,8)
    #pragma unroll
    for (int i = 0; i < 32; i += 8)
        tile[ty + i][tx] = in[(size_t)(kb + ty + i) * N + (nb + tx)];
    __syncthreads();
    #pragma unroll
    for (int i = 0; i < 32; i += 8)
        out[(size_t)(nb + ty + i) * K + (kb + tx)] = f2bf(tile[tx][ty + i]);
}

// ---------- fold BN (+conv bias) into per-channel scale/shift ----------
__global__ void prep_scale_shift_k(const float* __restrict__ cb, const float* __restrict__ g,
                                   const float* __restrict__ b,  const float* __restrict__ m,
                                   const float* __restrict__ v,  float* __restrict__ scale,
                                   float* __restrict__ shift, int C) {
    int c = threadIdx.x;
    if (c >= C) return;
    float s = g[c] * rsqrtf(v[c] + 1e-5f);
    scale[c] = s;
    shift[c] = (cb[c] - m[c]) * s + b[c];
}

// ---------- im2col (3x3, pad 1) bf16, K padded to Kp ----------
__global__ void im2col_bf16_k(const unsigned short* __restrict__ act, unsigned short* __restrict__ col,
                              int Cin, int H, int W, int K, int Kp) {
    int id = blockIdx.x * blockDim.x + threadIdx.x;
    int total = H * W * Kp;
    if (id >= total) return;
    int m = id / Kp, k = id % Kp;
    unsigned short v = 0;
    if (k < K) {
        int cin = k / 9, off = k % 9;
        int y = m / W + off / 3 - 1;
        int x = m % W + off % 3 - 1;
        if (y >= 0 && y < H && x >= 0 && x < W)
            v = act[(size_t)cin * H * W + (size_t)y * W + x];
    }
    col[(size_t)m * Kp + k] = v;
}

// ---------- WMMA bf16 GEMM: D = epilogue(A[MxK] * Bt[NxK]^T) ----------
// 128 threads = 4 waves; block tile 64(M) x 64(N); wave tile 32x32 (2x2 WMMA tiles).
// K staged 64 deep in LDS via CDNA5 async copies (global_load_async_to_lds_b128,
// tracked by ASYNCcnt, waited with s_wait_asynccnt) -- no VGPR round-trip.
// LDS row stride 72 halves = 144B: 16B-aligned rows (async b128 dest alignment)
// and 36-bank stride => conflict-free fragment reads across the 16 lanes.
// Epilogue: val = acc*scale[n] + shift[n]; optional relu; bf16 store
// (outRowMajor: D[m*N+n] for FC chain, else D[n*M+m] channel-major for convs).
#define LDT 72
__global__ __launch_bounds__(128) void gemm_wmma_bf16_k(
    const unsigned short* __restrict__ A,   // [M][K]
    const unsigned short* __restrict__ Bt,  // [N][K]
    unsigned short* __restrict__ D, int M, int N, int K,
    const float* __restrict__ scale, const float* __restrict__ shift,
    int relu, int outRowMajor) {

    __shared__ unsigned short sA[64 * LDT];
    __shared__ unsigned short sB[64 * LDT];

    const int tid  = threadIdx.x;
    const int lane = tid & 31;
    const int wv   = tid >> 5;          // 0..3
    const int wm   = (wv & 1) << 5;     // 0 or 32 (M half)
    const int wn   = (wv >> 1) << 5;    // 0 or 32 (N half)
    const int mBase = blockIdx.y * 64;
    const int nBase = blockIdx.x * 64;

    // staging map: 128 threads cover 64 rows x 64 halves; 4x async 16B copies per matrix
    const int lr = tid >> 1;            // row 0..63
    const int lk = (tid & 1) << 5;      // k-offset 0 or 32

    v8f acc[2][2] = {{{}, {}}, {{}, {}}};

    for (int k0 = 0; k0 < K; k0 += 64) {
        // async stage: memory -> LDS directly (ASYNCcnt path)
        #pragma unroll
        for (int j = 0; j < 4; ++j) {
            unsigned int la = (unsigned int)(size_t)(&sA[lr * LDT + lk + 8 * j]);
            unsigned long long gaddr_a =
                (unsigned long long)(size_t)(A + (size_t)(mBase + lr) * K + k0 + lk + 8 * j);
            asm volatile("global_load_async_to_lds_b128 %0, %1, off"
                         :: "v"(la), "v"(gaddr_a) : "memory");
            unsigned int lb = (unsigned int)(size_t)(&sB[lr * LDT + lk + 8 * j]);
            unsigned long long gaddr_b =
                (unsigned long long)(size_t)(Bt + (size_t)(nBase + lr) * K + k0 + lk + 8 * j);
            asm volatile("global_load_async_to_lds_b128 %0, %1, off"
                         :: "v"(lb), "v"(gaddr_b) : "memory");
        }
        if (k0 + 64 < K) {
            __builtin_prefetch(A  + (size_t)(mBase + lr) * K + k0 + 64 + lk, 0, 1);
            __builtin_prefetch(Bt + (size_t)(nBase + lr) * K + k0 + 64 + lk, 0, 1);
        }
        asm volatile("s_wait_asynccnt 0x0" ::: "memory");
        __syncthreads();

        // ISA 16-bit A/B fragment layout: lane L holds row (L&15);
        // lanes 0-15: K={0..7,16..23}+kk, lanes 16-31: K={8..15,24..31}+kk.
        const int lrow  = lane & 15;
        const int khalf = (lane >> 4) << 3;
        #pragma unroll
        for (int kk = 0; kk < 64; kk += 32) {
            FragBF16 fa[2], fb[2];
            #pragma unroll
            for (int i = 0; i < 8; ++i) {
                int k = kk + ((i < 4) ? (khalf + 2 * i) : (16 + khalf + 2 * (i - 4)));
                fa[0].u[i] = *reinterpret_cast<const unsigned int*>(&sA[(wm +      lrow) * LDT + k]);
                fa[1].u[i] = *reinterpret_cast<const unsigned int*>(&sA[(wm + 16 + lrow) * LDT + k]);
                fb[0].u[i] = *reinterpret_cast<const unsigned int*>(&sB[(wn +      lrow) * LDT + k]);
                fb[1].u[i] = *reinterpret_cast<const unsigned int*>(&sB[(wn + 16 + lrow) * LDT + k]);
            }
            #pragma unroll
            for (int i = 0; i < 2; ++i)
                #pragma unroll
                for (int j = 0; j < 2; ++j)
                    acc[i][j] = __builtin_amdgcn_wmma_f32_16x16x32_bf16(
                        false, fa[i].v, false, fb[j].v, (short)0, acc[i][j], false, false);
        }
        __syncthreads();
    }

    // Epilogue. C/D layout: VGPR r -> M = r + 8*(lane>=16), N = lane&15.
    const int lrow = lane & 15;
    const int msub = (lane >> 4) << 3;
    #pragma unroll
    for (int i = 0; i < 2; ++i) {
        #pragma unroll
        for (int j = 0; j < 2; ++j) {
            const int n  = nBase + wn + j * 16 + lrow;
            const float s  = scale ? scale[n] : 1.0f;
            const float sh = shift ? shift[n] : 0.0f;
            #pragma unroll
            for (int r = 0; r < 8; ++r) {
                float v = acc[i][j][r] * s + sh;
                if (relu) v = fmaxf(v, 0.0f);
                const int m = mBase + wm + i * 16 + msub + r;
                if (outRowMajor) D[(size_t)m * N + n] = f2bf(v);
                else             D[(size_t)n * M + m] = f2bf(v);
            }
        }
    }
}

// ---------- 2x2 maxpool, bf16 ----------
__global__ void maxpool2_bf16_k(const unsigned short* __restrict__ in, unsigned short* __restrict__ out,
                                int C, int H, int W) {
    int h2 = H >> 1, w2 = W >> 1;
    int id = blockIdx.x * blockDim.x + threadIdx.x;
    if (id >= C * h2 * w2) return;
    int c = id / (h2 * w2), rr = id % (h2 * w2);
    int y = rr / w2, x = rr % w2;
    const unsigned short* p = in + (size_t)c * H * W + (size_t)(2 * y) * W + 2 * x;
    float m = fmaxf(fmaxf(bf2f(p[0]), bf2f(p[1])), fmaxf(bf2f(p[W]), bf2f(p[W + 1])));
    out[id] = f2bf(m);
}

// ---------- SPP adaptive max pool over 64 ROIs, levels 1/2/3 ----------
__global__ void spp_pool_k(const unsigned short* __restrict__ feat, const int* __restrict__ boxes,
                           unsigned short* __restrict__ Y) {
    const int H = 32, W = 32;
    int id = blockIdx.x * blockDim.x + threadIdx.x;
    if (id >= 64 * 512 * 14) return;
    int r = id / (512 * 14), rem = id % (512 * 14);
    int c = rem / 14, cell = rem % 14;
    int n, i, j, off;
    if (cell == 0)      { n = 1; i = 0; j = 0; off = 0; }
    else if (cell < 5)  { n = 2; int q = cell - 1; i = q >> 1; j = q & 1; off = 512; }
    else                { n = 3; int q = cell - 5; i = q / 3;  j = q % 3; off = 2560; }
    const int* b = boxes + r * 4;
    int y0 = b[0], x0 = b[1], hh = b[2], ww = b[3];
    int rs = y0 + (i * hh) / n;
    int re = y0 + ((i + 1) * hh + n - 1) / n;   // ceil, matches y0 - (-(k+1)*hh)//n
    int cs = x0 + (j * ww) / n;
    int ce = x0 + ((j + 1) * ww + n - 1) / n;
    rs = max(rs, 0); cs = max(cs, 0); re = min(re, H); ce = min(ce, W);
    float m = -3.0e38f;
    const unsigned short* f = feat + (size_t)c * H * W;
    for (int y = rs; y < re; ++y)
        for (int x = cs; x < ce; ++x)
            m = fmaxf(m, bf2f(f[y * W + x]));
    Y[(size_t)r * 7168 + off + c * n * n + i * n + j] = f2bf(m);
}

// ---------- tiny classifiers: xc/xd = relu(h7 @ w + b), [64][20] ----------
__global__ void classifier_k(const unsigned short* __restrict__ h7,
                             const float* __restrict__ wc, const float* __restrict__ bc,
                             const float* __restrict__ wd, const float* __restrict__ bd,
                             float* __restrict__ xc, float* __restrict__ xd) {
    int id = blockIdx.x * blockDim.x + threadIdx.x;
    if (id >= 64 * 20) return;
    int r = id / 20, j = id % 20;
    float sc = 0.f, sd = 0.f;
    const unsigned short* h = h7 + (size_t)r * 4096;
    for (int k = 0; k < 4096; ++k) {
        float hv = bf2f(h[k]);
        sc += hv * wc[k * 20 + j];
        sd += hv * wd[k * 20 + j];
    }
    xc[id] = fmaxf(sc + bc[j], 0.f);
    xd[id] = fmaxf(sd + bd[j], 0.f);
}

__global__ void softmax_rows_k(const float* __restrict__ x, float* __restrict__ y) { // over 20 classes
    int r = threadIdx.x;
    if (r >= 64) return;
    float mx = -1e30f;
    for (int j = 0; j < 20; ++j) mx = fmaxf(mx, x[r * 20 + j]);
    float e[20], s = 0.f;
    for (int j = 0; j < 20; ++j) { e[j] = __expf(x[r * 20 + j] - mx); s += e[j]; }
    for (int j = 0; j < 20; ++j) y[r * 20 + j] = e[j] / s;
}

__global__ void softmax_cols_k(const float* __restrict__ x, float* __restrict__ y) { // over 64 rows
    int j = threadIdx.x;
    if (j >= 20) return;
    float mx = -1e30f;
    for (int r = 0; r < 64; ++r) mx = fmaxf(mx, x[r * 20 + j]);
    float s = 0.f;
    for (int r = 0; r < 64; ++r) s += __expf(x[r * 20 + j] - mx);
    for (int r = 0; r < 64; ++r) y[r * 20 + j] = __expf(x[r * 20 + j] - mx) / s;
}

// out layout: [0..19]=sum_r sc*sd ; [20..1299]=segma_d ; [1300..2579]=segma_c
__global__ void finalize_k(const float* __restrict__ sc, const float* __restrict__ sd,
                           float* __restrict__ out) {
    int id = blockIdx.x * blockDim.x + threadIdx.x;
    if (id >= 1280) return;
    out[20 + id]   = sd[id];
    out[1300 + id] = sc[id];
    if (id < 20) {
        float s = 0.f;
        for (int r = 0; r < 64; ++r) s += sc[r * 20 + id] * sd[r * 20 + id];
        out[id] = s;
    }
}

// ---------- host orchestration ----------
static inline unsigned nblk(long long n) { return (unsigned)((n + 255) / 256); }

extern "C" void kernel_launch(void* const* d_in, const int* in_sizes, int n_in,
                              void* d_out, int out_size, void* d_ws, size_t ws_size,
                              hipStream_t stream) {
    (void)in_sizes; (void)n_in; (void)out_size; (void)ws_size;
    // input indices: 0:x 1:ssw 2-9:conv_w 10-17:conv_b 18-25:bn_g 26-33:bn_b
    //                34-41:bn_m 42-49:bn_v 50:w6 51:b6 52:w7 53:b7 54:wc 55:bc 56:wd 57:bd
    char* ws = (char*)d_ws;
    size_t cur_off = 0;
    auto alloc = [&](size_t bytes) { size_t o = cur_off; cur_off = (cur_off + bytes + 255) & ~(size_t)255; return o; };

    const size_t ACT_BYTES = 64ull * 512 * 512 * 2;     // largest activation (conv1 out)
    const size_t COL_BYTES = 65536ull * 576 * 2;        // largest im2col; also >= w6/w7 bf16
    unsigned short* actA  = (unsigned short*)(ws + alloc(ACT_BYTES));
    unsigned short* actB  = (unsigned short*)(ws + alloc(ACT_BYTES));
    unsigned short* col   = (unsigned short*)(ws + alloc(COL_BYTES));
    unsigned short* wconv = (unsigned short*)(ws + alloc(4608ull * 512 * 2));
    float* scale = (float*)(ws + alloc(512 * 4));
    float* shift = (float*)(ws + alloc(512 * 4));
    unsigned short* Ybuf = (unsigned short*)(ws + alloc(64ull * 7168 * 2));
    unsigned short* h6   = (unsigned short*)(ws + alloc(64ull * 4096 * 2));
    unsigned short* h7   = (unsigned short*)(ws + alloc(64ull * 4096 * 2));
    float* xc = (float*)(ws + alloc(64 * 20 * 4));
    float* xd = (float*)(ws + alloc(64 * 20 * 4));
    float* sc = (float*)(ws + alloc(64 * 20 * 4));
    float* sd = (float*)(ws + alloc(64 * 20 * 4));

    // input image -> bf16
    {
        int n = 3 * 512 * 512;
        f32_to_bf16_k<<<nblk(n), 256, 0, stream>>>((const float*)d_in[0], actA, n);
    }

    struct ConvCfg { int cin, cout, h, w, pool_after; };
    const ConvCfg convs[8] = {
        {  3,  64, 512, 512, 1},
        { 64, 128, 256, 256, 1},
        {128, 256, 128, 128, 0},
        {256, 256, 128, 128, 1},
        {256, 512,  64,  64, 0},
        {512, 512,  64,  64, 1},
        {512, 512,  32,  32, 0},
        {512, 512,  32,  32, 0},
    };

    unsigned short* curp = actA;
    unsigned short* othp = actB;
    for (int i = 0; i < 8; ++i) {
        const ConvCfg& cc = convs[i];
        const int K  = cc.cin * 9;
        const int Kp = (K + 63) & ~63;      // pad to 64 for the 64-deep K stage
        const int N  = cc.cout;
        const int M  = cc.h * cc.w;

        conv_w_to_bf16_k<<<nblk((long long)Kp * N), 256, 0, stream>>>(
            (const float*)d_in[2 + i], wconv, K, Kp, N);
        prep_scale_shift_k<<<1, 512, 0, stream>>>(
            (const float*)d_in[10 + i], (const float*)d_in[18 + i], (const float*)d_in[26 + i],
            (const float*)d_in[34 + i], (const float*)d_in[42 + i], scale, shift, N);
        im2col_bf16_k<<<nblk((long long)M * Kp), 256, 0, stream>>>(
            curp, col, cc.cin, cc.h, cc.w, K, Kp);

        dim3 g(N / 64, M / 64);
        gemm_wmma_bf16_k<<<g, 128, 0, stream>>>(col, wconv, othp, M, N, Kp,
                                                scale, shift, /*relu=*/1, /*rowMajor=*/0);
        { unsigned short* t = curp; curp = othp; othp = t; }

        if (cc.pool_after) {
            int h2 = cc.h / 2, w2 = cc.w / 2;
            maxpool2_bf16_k<<<nblk((long long)N * h2 * w2), 256, 0, stream>>>(curp, othp, N, cc.h, cc.w);
            { unsigned short* t = curp; curp = othp; othp = t; }
        }
    }

    // SPP pool: feat [512,32,32] bf16 -> Y [64,7168] bf16
    spp_pool_k<<<nblk(64LL * 512 * 14), 256, 0, stream>>>(curp, (const int*)d_in[1], Ybuf);

    // FC6: [64,7168] @ [7168,4096] + b6, relu (w6 transposed to [4096][7168] bf16)
    {
        dim3 tb(32, 8), tg(7168 / 32, 4096 / 32);
        transpose_f32_to_bf16_k<<<tg, tb, 0, stream>>>((const float*)d_in[50], col, 7168, 4096);
        dim3 g(4096 / 64, 64 / 64);
        gemm_wmma_bf16_k<<<g, 128, 0, stream>>>(Ybuf, col, h6, 64, 4096, 7168,
                                                nullptr, (const float*)d_in[51], 1, 1);
    }
    // FC7: [64,4096] @ [4096,4096] + b7, relu
    {
        dim3 tb(32, 8), tg(4096 / 32, 4096 / 32);
        transpose_f32_to_bf16_k<<<tg, tb, 0, stream>>>((const float*)d_in[52], col, 4096, 4096);
        dim3 g(4096 / 64, 64 / 64);
        gemm_wmma_bf16_k<<<g, 128, 0, stream>>>(h6, col, h7, 64, 4096, 4096,
                                                nullptr, (const float*)d_in[53], 1, 1);
    }

    classifier_k<<<nblk(64 * 20), 256, 0, stream>>>(
        h7, (const float*)d_in[54], (const float*)d_in[55],
        (const float*)d_in[56], (const float*)d_in[57], xc, xd);
    softmax_rows_k<<<1, 64, 0, stream>>>(xc, sc);
    softmax_cols_k<<<1, 32, 0, stream>>>(xd, sd);
    finalize_k<<<nblk(1280), 256, 0, stream>>>(sc, sd, (float*)d_out);
}
